// ParallelTransformerBlock_28054726377505
// MI455X (gfx1250) — matrix operand order repaired and checked by reading, not compiled, not run on hardware
//
#include <hip/hip_runtime.h>
#include <hip/hip_bf16.h>
#include <math.h>

typedef __bf16 bhalf;
typedef bhalf v16bf __attribute__((ext_vector_type(16)));
typedef float v8f  __attribute__((ext_vector_type(8)));

#define DIM_   2048
#define HEADS_ 16
#define DH_    64
#define AI_    1024      // ATTN_INNER
#define FFI_   8192
#define NF_    17536     // fused width = 1024 + 64 + 64 + 2*8192
#define MROWS_ 4096      // B*N
#define SEQ_   2048

static __device__ __forceinline__ bhalf f2bf(float f) { return (bhalf)f; }

// load a v16bf fragment from two contiguous 16-byte chunks
static __device__ __forceinline__ v16bf ld_frag(const bhalf* p0, const bhalf* p1) {
  union { uint4 u[2]; v16bf v; } t;
  t.u[0] = *(const uint4*)p0;
  t.u[1] = *(const uint4*)p1;
  return t.v;
}

// ---------------- async global->LDS copy (CDNA5 GLOBAL_LOAD_ASYNC_TO_LDS_B128)
#if defined(__has_builtin)
#if __has_builtin(__builtin_amdgcn_global_load_async_to_lds_b128) && \
    __has_builtin(__builtin_amdgcn_s_wait_asynccnt)
#define HAVE_ASYNC_COPY 1
#endif
#endif
#ifndef HAVE_ASYNC_COPY
#define HAVE_ASYNC_COPY 0
#endif

#if HAVE_ASYNC_COPY
// ROCm declares the builtin as taking generic pointers to 16-byte int vectors
// (diagnostic: "__attribute__((__vector_size__(4 * sizeof(int)))) int *").
typedef int vec4i_ __attribute__((vector_size(16)));
static __device__ __forceinline__ void async_cp16(const void* g, void* l) {
  __builtin_amdgcn_global_load_async_to_lds_b128((vec4i_*)g, (vec4i_*)l, 0, 0);
}
static __device__ __forceinline__ void async_wait() {
  __builtin_amdgcn_s_wait_asynccnt(0);
}
#define STAGE16(gptr, lptr) async_cp16((const void*)(gptr), (void*)(lptr))
#define STAGE_FENCE()       async_wait()
#else
#define STAGE16(gptr, lptr) (*(uint4*)(lptr) = *(const uint4*)(gptr))
#define STAGE_FENCE()       ((void)0)
#endif

// ---------------------------------------------------------------- layernorm
__global__ __launch_bounds__(256)
void ln_kernel(const float* __restrict__ x, const float* __restrict__ gamma,
               bhalf* __restrict__ xnb) {
  int row = blockIdx.x;
  const float* xr = x + (size_t)row * DIM_;
  __shared__ float red[256];
  float s = 0.f, ss = 0.f;
  for (int c = threadIdx.x; c < DIM_; c += 256) { float v = xr[c]; s += v; ss += v * v; }
  red[threadIdx.x] = s; __syncthreads();
  for (int st = 128; st > 0; st >>= 1) { if (threadIdx.x < st) red[threadIdx.x] += red[threadIdx.x + st]; __syncthreads(); }
  float mean = red[0] * (1.f / DIM_);
  __syncthreads();
  red[threadIdx.x] = ss; __syncthreads();
  for (int st = 128; st > 0; st >>= 1) { if (threadIdx.x < st) red[threadIdx.x] += red[threadIdx.x + st]; __syncthreads(); }
  float var = red[0] * (1.f / DIM_) - mean * mean;
  float rstd = rsqrtf(var + 1e-5f);
  for (int c = threadIdx.x; c < DIM_; c += 256)
    xnb[(size_t)row * DIM_ + c] = f2bf((xr[c] - mean) * rstd * gamma[c]);
}

// ---------------------------------------------------------------- cast + transpose: f32 [K][N] -> bf16 [N][K]
__global__ __launch_bounds__(256)
void castT_kernel(const float* __restrict__ in, bhalf* __restrict__ out, int K, int N) {
  __shared__ float t[32][33];
  int n0 = blockIdx.x * 32, k0 = blockIdx.y * 32;
  int tx = threadIdx.x, ty = threadIdx.y;   // 32 x 8
#pragma unroll
  for (int i = 0; i < 4; ++i)
    t[ty + 8 * i][tx] = in[(size_t)(k0 + ty + 8 * i) * N + n0 + tx];
  __syncthreads();
#pragma unroll
  for (int i = 0; i < 4; ++i)
    out[(size_t)(n0 + ty + 8 * i) * K + k0 + tx] = f2bf(t[tx][ty + 8 * i]);
}

// ---------------------------------------------------------------- WMMA GEMM
// C[M,N](f32) = A[M,K](bf16, lda) @ W[K,N] given as Bt = W^T (bf16 [N][K], ldbt)
// block 256 thr / 8 waves; tile BM=128 x BN=128, BK=32; wave -> 32x64 (2x4 wmma)
#define BM 128
#define BN 128
#define BK 32
#define LDT 40   // padded LDS row (bf16 elems): 80B, 16B-aligned, bank-stride 20
template <bool ACC>
__global__ __launch_bounds__(256)
void gemm_kernel(const bhalf* __restrict__ A, const bhalf* __restrict__ Bt,
                 float* __restrict__ C, int Kdim, int lda, int ldbt, int ldc) {
  __shared__ __align__(16) bhalf As[BM][LDT];
  __shared__ __align__(16) bhalf Bs[BN][LDT];
  int tid = threadIdx.x;
  int lane = tid & 31, wave = tid >> 5;
  int c15 = lane & 15, kh = lane >> 4;
  int bm = blockIdx.y * BM, bn = blockIdx.x * BN;
  int wm = (wave >> 1) * 32, wn = (wave & 1) * 64;
  v8f acc[2][4] = {};
  for (int k0 = 0; k0 < Kdim; k0 += BK) {
    // stage A (128x32) and B^T (128x32): 512 16B-chunks each, 2/thread
#pragma unroll
    for (int i = 0; i < 2; ++i) {
      int c = tid + i * 256;
      int r = c >> 2, col = (c & 3) << 3;
      STAGE16(&A[(size_t)(bm + r) * lda + k0 + col], &As[r][col]);
      STAGE16(&Bt[(size_t)(bn + r) * ldbt + k0 + col], &Bs[r][col]);
    }
    if (k0 + BK < Kdim) {
      __builtin_prefetch(&A[(size_t)(bm + (tid >> 1)) * lda + k0 + BK], 0, 1);
      __builtin_prefetch(&Bt[(size_t)(bn + (tid >> 1)) * ldbt + k0 + BK], 0, 1);
    }
    STAGE_FENCE();
    __syncthreads();
    v16bf af[2];
#pragma unroll
    for (int i = 0; i < 2; ++i) {
      const bhalf* rp = &As[wm + i * 16 + c15][0];
      af[i] = ld_frag(rp + kh * 8, rp + 16 + kh * 8);     // A layout: K = kh*8+e, 16+kh*8+e
    }
#pragma unroll
    for (int j = 0; j < 4; ++j) {
      const bhalf* rp = &Bs[wn + j * 16 + c15][0];
      v16bf bfr = ld_frag(rp + kh * 8, rp + 16 + kh * 8);  // B layout: K = kh*8+e, 16+kh*8+e
#pragma unroll
      for (int i = 0; i < 2; ++i)
        acc[i][j] = __builtin_amdgcn_wmma_f32_16x16x32_bf16(
            false, af[i], false, bfr, (short)0, acc[i][j], false, false);
    }
    __syncthreads();
  }
#pragma unroll
  for (int i = 0; i < 2; ++i)
#pragma unroll
    for (int j = 0; j < 4; ++j) {
      int rbase = bm + wm + i * 16 + kh * 8;
      int cidx  = bn + wn + j * 16 + c15;
#pragma unroll
      for (int r = 0; r < 8; ++r) {
        size_t idx = (size_t)(rbase + r) * ldc + cidx;
        if (ACC) C[idx] += acc[i][j][r]; else C[idx] = acc[i][j][r];
      }
    }
}

// ---------------------------------------------------------------- LoRA t = in @ a   (in: bf16 M x K, a: f32 K x 8)
__global__ __launch_bounds__(256)
void lora_t_kernel(const bhalf* __restrict__ inp, int ldin, int K,
                   const float* __restrict__ a, float* __restrict__ t) {
  int row = blockIdx.x;
  int j = threadIdx.x & 7, part = threadIdx.x >> 3;   // 32 parts
  int kper = K / 32;
  const bhalf* ir = inp + (size_t)row * ldin;
  float s = 0.f;
  for (int k = part * kper; k < (part + 1) * kper; ++k) s += (float)ir[k] * a[k * 8 + j];
  __shared__ float red[256];
  red[threadIdx.x] = s; __syncthreads();
  for (int st = 128; st >= 8; st >>= 1) { if (threadIdx.x < st) red[threadIdx.x] += red[threadIdx.x + st]; __syncthreads(); }
  if (threadIdx.x < 8) t[row * 8 + threadIdx.x] = red[threadIdx.x];
}

// ---------------------------------------------------------------- epilogue: LoRA add + scale + RoPE + SiLU
// writes q (bf16 [row][1024]), k (bf16 [row][64]), V^T (bf16 [b][64][2048]), ff act
__global__ __launch_bounds__(256)
void epi_kernel(const float* __restrict__ C,
                const float* __restrict__ tq, const float* __restrict__ tk, const float* __restrict__ tv,
                const float* __restrict__ bq, const float* __restrict__ bk, const float* __restrict__ bv,
                bhalf* __restrict__ qr, bhalf* __restrict__ kr, bhalf* __restrict__ vt,
                bhalf* __restrict__ ffa) {
  int row = blockIdx.x;
  int b = row >> 11, n = row & (SEQ_ - 1);
  const float* Cr = C + (size_t)row * NF_;
  __shared__ float qs[AI_];
  __shared__ float ks[DH_];
  float tqv[8], tkv[8], tvv[8];
#pragma unroll
  for (int j = 0; j < 8; ++j) { tqv[j] = tq[row*8+j]; tkv[j] = tk[row*8+j]; tvv[j] = tv[row*8+j]; }
  for (int c = threadIdx.x; c < AI_; c += 256) {
    float v = Cr[c];
#pragma unroll
    for (int j = 0; j < 8; ++j) v += tqv[j] * bq[j * AI_ + c];
    qs[c] = v * 0.125f;                           // DIM_HEAD^-0.5
  }
  for (int c = threadIdx.x; c < DH_; c += 256) {
    float v = Cr[AI_ + c];
#pragma unroll
    for (int j = 0; j < 8; ++j) v += tkv[j] * bk[j * DH_ + c];
    ks[c] = v;
    float w = Cr[AI_ + DH_ + c];
#pragma unroll
    for (int j = 0; j < 8; ++j) w += tvv[j] * bv[j * DH_ + c];
    vt[((size_t)(b * DH_ + c)) * SEQ_ + n] = f2bf(w);   // transposed V
  }
  __syncthreads();
  for (int c = threadIdx.x; c < AI_; c += 256) {
    int d = c & 63, jj = d & 31, h = c >> 6;
    float inv = powf(10000.f, -(float)(2 * jj) / 64.f);
    float th = (float)n * inv;
    float cs = __cosf(th), sn = __sinf(th);
    float other = (d < 32) ? -qs[h * 64 + d + 32] : qs[h * 64 + d - 32];
    qr[(size_t)row * AI_ + c] = f2bf(qs[c] * cs + other * sn);
  }
  for (int c = threadIdx.x; c < DH_; c += 256) {
    int jj = c & 31;
    float inv = powf(10000.f, -(float)(2 * jj) / 64.f);
    float th = (float)n * inv;
    float cs = __cosf(th), sn = __sinf(th);
    float other = (c < 32) ? -ks[c + 32] : ks[c - 32];
    kr[(size_t)row * DH_ + c] = f2bf(ks[c] * cs + other * sn);
  }
  const float* fx = Cr + AI_ + 2 * DH_;
  const float* fg = fx + FFI_;
  for (int c = threadIdx.x; c < FFI_; c += 256) {
    float g = fg[c];
    float act = g / (1.f + __expf(-g));           // silu
    ffa[(size_t)row * FFI_ + c] = f2bf(act * fx[c]);
  }
}

// ---------------------------------------------------------------- flash attention (causal, single KV head)
// grid.x = B*HEADS (32), grid.y = SEQ/64 (32); block 128 (4 waves, 16 q-rows each)
#define LDK 72   // Ks padded row (144B)
#define LDV 40   // Vst / Ps padded row (80B)
__global__ __launch_bounds__(128)
void attn_kernel(const bhalf* __restrict__ qr, const bhalf* __restrict__ kr,
                 const bhalf* __restrict__ vt, bhalf* __restrict__ ao) {
  int bh = blockIdx.x, b = bh >> 4, h = bh & 15;
  int qt = blockIdx.y;
  int tid = threadIdx.x, lane = tid & 31, wave = tid >> 5;
  int c15 = lane & 15, kh = lane >> 4;
  int qrow = qt * 64 + wave * 16 + c15;
  const bhalf* qp = qr + ((size_t)(b * SEQ_ + qrow)) * AI_ + h * DH_;
  v16bf qf[2];
#pragma unroll
  for (int f = 0; f < 2; ++f)
    qf[f] = ld_frag(qp + f * 32 + kh * 8, qp + f * 32 + 16 + kh * 8);
  __shared__ __align__(16) bhalf Ks[32][LDK];     // [key][d]
  __shared__ __align__(16) bhalf Vst[DH_][LDV];   // [d][key]
  __shared__ __align__(16) bhalf Ps[4][16][LDV];  // per-wave P tile
  v8f o[4] = {};
  float mval[8], lsum[8];
#pragma unroll
  for (int r = 0; r < 8; ++r) { mval[r] = -3.0e38f; lsum[r] = 0.f; }
  int jend = qt * 64 + 64;
  for (int j0 = 0; j0 < jend; j0 += 32) {
    // stage K (32x64, key-major) + V^T (64 d x 32 key): 256 16B-chunks each, 2/thread
#pragma unroll
    for (int i = 0; i < 2; ++i) {
      int c = tid + i * 128;
      int r = c >> 3, col = (c & 7) << 3;
      STAGE16(&kr[((size_t)(b * SEQ_ + j0 + r)) * DH_ + col], &Ks[r][col]);
      int d = c >> 2, koff = (c & 3) << 3;
      STAGE16(&vt[((size_t)(b * DH_ + d)) * SEQ_ + j0 + koff], &Vst[d][koff]);
    }
    STAGE_FENCE();
    __syncthreads();
    v8f s[2] = {};
#pragma unroll
    for (int jt = 0; jt < 2; ++jt)
#pragma unroll
      for (int f = 0; f < 2; ++f) {
        const bhalf* rp = &Ks[jt * 16 + c15][f * 32 + kh * 8];  // B[k][n]=K[n][d]
        v16bf bfr = ld_frag(rp, rp + 16);
        s[jt] = __builtin_amdgcn_wmma_f32_16x16x32_bf16(
            false, qf[f], false, bfr, (short)0, s[jt], false, false);
      }
    int row0 = qt * 64 + wave * 16 + kh * 8;
#pragma unroll
    for (int r = 0; r < 8; ++r) {
      int irow = row0 + r;
      float mx = mval[r];
#pragma unroll
      for (int jt = 0; jt < 2; ++jt) {
        int col = j0 + jt * 16 + c15;
        if (col > irow) s[jt][r] = -3.0e38f;      // causal mask
        mx = fmaxf(mx, s[jt][r]);
      }
      for (int m = 1; m < 16; m <<= 1) mx = fmaxf(mx, __shfl_xor(mx, m, 32));
      float scale = __expf(mval[r] - mx);
      float ssum = 0.f;
#pragma unroll
      for (int jt = 0; jt < 2; ++jt) {
        float p = __expf(s[jt][r] - mx);
        s[jt][r] = p; ssum += p;
      }
      for (int m = 1; m < 16; m <<= 1) ssum += __shfl_xor(ssum, m, 32);
      lsum[r] = lsum[r] * scale + ssum;
      mval[r] = mx;
#pragma unroll
      for (int ot = 0; ot < 4; ++ot) o[ot][r] *= scale;
    }
#pragma unroll
    for (int jt = 0; jt < 2; ++jt)
#pragma unroll
      for (int r = 0; r < 8; ++r)
        Ps[wave][kh * 8 + r][jt * 16 + c15] = f2bf(s[jt][r]);
    __syncthreads();
    const bhalf* pp = &Ps[wave][c15][0];
    v16bf pf = ld_frag(pp + kh * 8, pp + 16 + kh * 8);
#pragma unroll
    for (int ot = 0; ot < 4; ++ot) {
      const bhalf* rp = &Vst[ot * 16 + c15][kh * 8];            // B[k][n]=V[k][d]
      v16bf vfr = ld_frag(rp, rp + 16);
      o[ot] = __builtin_amdgcn_wmma_f32_16x16x32_bf16(
          false, pf, false, vfr, (short)0, o[ot], false, false);
    }
    __syncthreads();
  }
#pragma unroll
  for (int ot = 0; ot < 4; ++ot)
#pragma unroll
    for (int r = 0; r < 8; ++r) {
      int irow = qt * 64 + wave * 16 + kh * 8 + r;
      float val = o[ot][r] / lsum[r];
      ao[((size_t)(b * SEQ_ + irow)) * AI_ + h * DH_ + ot * 16 + c15] = f2bf(val);
    }
}

// ---------------------------------------------------------------- out += t_o @ b_o
__global__ __launch_bounds__(256)
void lora_out_kernel(const float* __restrict__ t, const float* __restrict__ bo,
                     float* __restrict__ out) {
  size_t i = (size_t)blockIdx.x * 256 + threadIdx.x;   // MROWS_*DIM_ elems
  int row = (int)(i >> 11), c = (int)(i & (DIM_ - 1));
  float s = out[i];
#pragma unroll
  for (int j = 0; j < 8; ++j) s += t[row * 8 + j] * bo[j * DIM_ + c];
  out[i] = s;
}

// ---------------------------------------------------------------- launch
extern "C" void kernel_launch(void* const* d_in, const int* in_sizes, int n_in,
                              void* d_out, int out_size, void* d_ws, size_t ws_size,
                              hipStream_t stream) {
  const float* x      = (const float*)d_in[0];
  const float* gamma  = (const float*)d_in[1];
  const float* wfused = (const float*)d_in[2];
  const float* wattn  = (const float*)d_in[3];
  const float* wff    = (const float*)d_in[4];
  const float* a_q = (const float*)d_in[5],  *b_q = (const float*)d_in[6];
  const float* a_k = (const float*)d_in[7],  *b_k = (const float*)d_in[8];
  const float* a_v = (const float*)d_in[9],  *b_v = (const float*)d_in[10];
  const float* a_o = (const float*)d_in[11], *b_o = (const float*)d_in[12];
  float* out = (float*)d_out;

  char* ws = (char*)d_ws;
  size_t off = 0;
  auto alloc = [&](size_t bytes) { size_t o = off; off += (bytes + 255) & ~(size_t)255; return o; };
  bhalf* xnb   = (bhalf*)(ws + alloc((size_t)MROWS_ * DIM_ * 2));
  bhalf* wfT   = (bhalf*)(ws + alloc((size_t)DIM_ * NF_ * 2));   // [NF][DIM]
  bhalf* waT   = (bhalf*)(ws + alloc((size_t)AI_ * DIM_ * 2));   // [DIM][AI]
  bhalf* wffT  = (bhalf*)(ws + alloc((size_t)FFI_ * DIM_ * 2));  // [DIM][FFI]
  float* Cf    = (float*)(ws + alloc((size_t)MROWS_ * NF_ * 4));
  bhalf* qrb   = (bhalf*)(ws + alloc((size_t)MROWS_ * AI_ * 2));
  bhalf* krb   = (bhalf*)(ws + alloc((size_t)MROWS_ * DH_ * 2));
  bhalf* vtb   = (bhalf*)(ws + alloc((size_t)MROWS_ * DH_ * 2)); // [B][64][SEQ]
  bhalf* ffa   = (bhalf*)(ws + alloc((size_t)MROWS_ * FFI_ * 2));
  bhalf* attno = (bhalf*)(ws + alloc((size_t)MROWS_ * AI_ * 2));
  float* tq    = (float*)(ws + alloc((size_t)MROWS_ * 8 * 4));
  float* tk    = (float*)(ws + alloc((size_t)MROWS_ * 8 * 4));
  float* tv    = (float*)(ws + alloc((size_t)MROWS_ * 8 * 4));
  float* to    = (float*)(ws + alloc((size_t)MROWS_ * 8 * 4));

  // 1. layernorm -> bf16
  ln_kernel<<<MROWS_, 256, 0, stream>>>(x, gamma, xnb);
  // 2. weight casts with transpose (W[K][N] -> bf16 W^T[N][K])
  castT_kernel<<<dim3(NF_ / 32, DIM_ / 32), dim3(32, 8), 0, stream>>>(wfused, wfT, DIM_, NF_);
  castT_kernel<<<dim3(DIM_ / 32, AI_ / 32), dim3(32, 8), 0, stream>>>(wattn, waT, AI_, DIM_);
  castT_kernel<<<dim3(DIM_ / 32, FFI_ / 32), dim3(32, 8), 0, stream>>>(wff, wffT, FFI_, DIM_);
  // 3. fused GEMM: Cf = xn @ w_fused
  gemm_kernel<false><<<dim3(NF_ / BN, MROWS_ / BM), 256, 0, stream>>>(
      xnb, wfT, Cf, DIM_, DIM_, DIM_, NF_);
  // 4. LoRA t-vectors
  lora_t_kernel<<<MROWS_, 256, 0, stream>>>(xnb, DIM_, DIM_, a_q, tq);
  lora_t_kernel<<<MROWS_, 256, 0, stream>>>(xnb, DIM_, DIM_, a_k, tk);
  lora_t_kernel<<<MROWS_, 256, 0, stream>>>(xnb, DIM_, DIM_, a_v, tv);
  // 5. epilogue: lora add + rope + silu -> bf16 q/k/v^T/ff
  epi_kernel<<<MROWS_, 256, 0, stream>>>(Cf, tq, tk, tv, b_q, b_k, b_v,
                                         qrb, krb, vtb, ffa);
  // 6. causal flash attention
  attn_kernel<<<dim3(HEADS_ * 2, SEQ_ / 64), 128, 0, stream>>>(qrb, krb, vtb, attno);
  // 7. out = attno @ w_attn_out
  gemm_kernel<false><<<dim3(DIM_ / BN, MROWS_ / BM), 256, 0, stream>>>(
      attno, waT, out, AI_, AI_, AI_, DIM_);
  // 8. out += attno @ a_o @ b_o
  lora_t_kernel<<<MROWS_, 256, 0, stream>>>(attno, AI_, AI_, a_o, to);
  lora_out_kernel<<<(MROWS_ * DIM_) / 256, 256, 0, stream>>>(to, b_o, out);
  // 9. out += silu_ff @ w_ff_out
  gemm_kernel<true><<<dim3(DIM_ / BN, MROWS_ / BM), 256, 0, stream>>>(
      ffa, wffT, out, FFI_, FFI_, FFI_, DIM_);
  (void)in_sizes; (void)n_in; (void)out_size; (void)ws_size;
}
